// ClusterCenterEstimation_40638980554864
// MI455X (gfx1250) — matrix operand order repaired and checked by reading, not compiled, not run on hardware
//
#include <hip/hip_runtime.h>
#include <hip/hip_bf16.h>
#include <cstdint>
#include <cstddef>

// ---------------- problem constants ----------------
#define BS     8
#define C_IN   1024
#define H_F    48
#define W_F    160
#define HP     50          // padded H (halo 1)
#define WP     162         // padded W (halo 1)
#define OC     256
#define HW     (H_F * W_F)         // 7680
#define M_TOT  (BS * HW)           // 61440
#define TOK_H  12
#define TOK_W  40
#define N_TOK  (TOK_H * TOK_W)     // 480
#define K_CLUST 100
#define GN_EPS  1e-5f

typedef __attribute__((ext_vector_type(16))) _Float16 v16h;
typedef __attribute__((ext_vector_type(8)))  _Float16 v8h;
typedef __attribute__((ext_vector_type(8)))  float    v8f;

union V16 { v16h v; v8h h[2]; };

__device__ __forceinline__ v8h ld8(const _Float16* p) {
    return *reinterpret_cast<const v8h*>(p);
}

// ---------------- 1) pad + f32->f16, NCHW -> padded NHWC ----------------
__global__ __launch_bounds__(256)
void pad_f16_kernel(const float* __restrict__ feat, _Float16* __restrict__ fpad) {
    long long idx = (long long)blockIdx.x * 256 + threadIdx.x;
    const long long total = (long long)BS * HP * WP * C_IN;
    if (idx >= total) return;
    int c = (int)(idx & (C_IN - 1));
    long long t = idx >> 10;
    int wp = (int)(t % WP); t /= WP;
    int hp = (int)(t % HP);
    int b  = (int)(t / HP);
    float v = 0.0f;
    if (hp >= 1 && hp <= H_F && wp >= 1 && wp <= W_F) {
        v = feat[(((size_t)b * C_IN + c) * H_F + (hp - 1)) * W_F + (wp - 1)];
    }
    fpad[idx] = (_Float16)v;
}

// ---------------- 2) W1 OIHW -> f16 [o][tap*1024+i] ----------------
__global__ __launch_bounds__(256)
void wconv_f16_kernel(const float* __restrict__ W1, _Float16* __restrict__ wgt) {
    int idx = blockIdx.x * 256 + threadIdx.x;
    if (idx >= 9 * C_IN * OC) return;
    int o   = idx & 255;
    int t   = idx >> 8;
    int i   = t & 1023;
    int tap = t >> 10;
    wgt[(size_t)o * (9 * C_IN) + tap * C_IN + i] = (_Float16)W1[((size_t)o * C_IN + i) * 9 + tap];
}

// ---------------- 3) implicit-GEMM 3x3 conv via WMMA f16 ----------------
// block = 128 threads (4 waves). Block M-tile = 32 rows; wave owns N cols
// [w*64, w*64+64). Per wave: 2 A-tiles x 4 B-tiles = 8 accumulators, so each
// k-step issues 8 WMMA against 12x16B loads (1.5 loads/WMMA) and each B tile
// feeds two back-to-back WMMAs.
__global__ __launch_bounds__(128)
void conv3x3_wmma_kernel(const _Float16* __restrict__ fpad,
                         const _Float16* __restrict__ wgt,
                         const float*    __restrict__ b1,
                         float*          __restrict__ out) {
    const int lane  = threadIdx.x & 31;
    const int wave  = threadIdx.x >> 5;
    const int hi    = lane >> 4;       // half-wave selector (ISA A/B K layout)
    const int mrow  = lane & 15;
    const int mbase = blockIdx.x * 32;

    // per-lane im2col element offsets (fit in u32: padded tensor = 66.4M elems)
    unsigned ab[2][9];
#pragma unroll
    for (int s = 0; s < 2; ++s) {
        const int m   = mbase + s * 16 + mrow;
        const int b   = m / HW;
        const int rem = m - b * HW;
        const int h   = rem / W_F;
        const int w   = rem - h * W_F;
#pragma unroll
        for (int tap = 0; tap < 9; ++tap) {
            const int kh = tap / 3, kw = tap % 3;
            ab[s][tap] = (unsigned)(((b * HP + h + kh) * WP + (w + kw)) * C_IN + hi * 8);
        }
    }

    const int n0 = wave * 64;
    v8f acc[2][4];
#pragma unroll
    for (int s = 0; s < 2; ++s)
#pragma unroll
        for (int t = 0; t < 4; ++t) acc[s][t] = (v8f){};

    for (int tap = 0; tap < 9; ++tap) {
        const _Float16* a0  = fpad + ab[0][tap];
        const _Float16* a1  = fpad + ab[1][tap];
        // B column for this lane: N = n0 + (lane&15), K = tap*1024 + c0 + {hi*8..+7, 16+hi*8..+7}
        const _Float16* bp0 = wgt + (size_t)(n0 + (lane & 15)) * (9 * C_IN) + tap * C_IN + hi * 8;
        for (int cs = 0; cs < 32; ++cs) {
            const int c0 = cs * 32;
            // A 16x32 f16: lanes<16 hold K {0..7,16..23}, lanes>=16 {8..15,24..31}
            V16 A0, A1;
            A0.h[0] = ld8(a0 + c0); A0.h[1] = ld8(a0 + c0 + 16);
            A1.h[0] = ld8(a1 + c0); A1.h[1] = ld8(a1 + c0 + 16);
            const _Float16* bp = bp0 + c0;
#pragma unroll
            for (int t = 0; t < 4; ++t) {
                V16 B;
                B.h[0] = ld8(bp + (size_t)t * 16 * (9 * C_IN));
                B.h[1] = ld8(bp + (size_t)t * 16 * (9 * C_IN) + 16);
                acc[0][t] = __builtin_amdgcn_wmma_f32_16x16x32_f16(false, A0.v, false, B.v, (short)0, acc[0][t], false, false);
                acc[1][t] = __builtin_amdgcn_wmma_f32_16x16x32_f16(false, A1.v, false, B.v, (short)0, acc[1][t], false, false);
            }
        }
    }

    // C/D f32 layout: VGPR r -> M = tilebase + r + 8*hi, N = n0 + (lane&15)
    const int col = lane & 15;
#pragma unroll
    for (int s = 0; s < 2; ++s)
#pragma unroll
        for (int t = 0; t < 4; ++t)
#pragma unroll
            for (int r = 0; r < 8; ++r) {
                const int mm = mbase + s * 16 + r + hi * 8;
                out[(size_t)mm * OC + n0 + t * 16 + col] =
                    acc[s][t][r] + b1[n0 + t * 16 + col];
            }
}

// ---------------- 4) GroupNorm statistics (per batch, per group of 8 ch) ----------------
__global__ __launch_bounds__(256)
void gn_stats_kernel(const float* __restrict__ conv, float* __restrict__ stats) {
    const int b = blockIdx.x >> 5;
    const int g = blockIdx.x & 31;
    float s = 0.f, s2 = 0.f;
    for (int idx = threadIdx.x; idx < HW * 8; idx += 256) {
        const int pl = idx >> 3;
        const int ch = (g << 3) + (idx & 7);
        const float x = conv[((size_t)(b * HW + pl)) * OC + ch];
        s += x; s2 += x * x;
    }
    __shared__ float sh[256], sh2[256];
    sh[threadIdx.x] = s; sh2[threadIdx.x] = s2;
    __syncthreads();
    for (int o = 128; o; o >>= 1) {
        if (threadIdx.x < o) {
            sh[threadIdx.x]  += sh[threadIdx.x + o];
            sh2[threadIdx.x] += sh2[threadIdx.x + o];
        }
        __syncthreads();
    }
    if (threadIdx.x == 0) {
        const float inv  = 1.0f / (float)(HW * 8);
        const float mean = sh[0] * inv;
        const float var  = sh2[0] * inv - mean * mean;
        stats[2 * (b * 32 + g)]     = mean;
        stats[2 * (b * 32 + g) + 1] = rsqrtf(var + GN_EPS);
    }
}

// ---------------- 5) fused GN -> ReLU -> conv1x1 -> sigmoid -> +depth ----------------
// One wave per pixel; lane L handles channels [L*8, L*8+8) == group L.
__global__ __launch_bounds__(256)
void combined_kernel(const float* __restrict__ conv,
                     const float* __restrict__ stats,
                     const float* __restrict__ gamma,
                     const float* __restrict__ beta,
                     const float* __restrict__ W2,
                     const float* __restrict__ b2,
                     const float* __restrict__ calibs,
                     float* __restrict__ out_scores) {
    const int gid  = blockIdx.x * blockDim.x + threadIdx.x;
    const int p    = gid >> 5;
    const int lane = gid & 31;
    if (p >= M_TOT) return;
    const int b   = p / HW;
    const int rem = p - b * HW;
    const int h   = rem / W_F;

    const float mean = stats[2 * (b * 32 + lane)];
    const float rstd = stats[2 * (b * 32 + lane) + 1];
    float dot = 0.f;
#pragma unroll
    for (int j = 0; j < 8; ++j) {
        const int ch = lane * 8 + j;
        const float x = conv[(size_t)p * OC + ch];
        float y = (x - mean) * rstd * gamma[ch] + beta[ch];
        y = fmaxf(y, 0.f);
        dot += y * W2[ch];
    }
#pragma unroll
    for (int off = 16; off > 0; off >>= 1) dot += __shfl_xor(dot, off, 32);

    if (lane == 0) {
        const float hm = 1.0f / (1.0f + __expf(-(dot + b2[0])));
        const float fy  = calibs[b * 12 + 5];
        const float cyc = (float)H_F * calibs[b * 12 + 6] / 375.0f;
        const float ds  = -fmaxf(500.0f * ((float)h - cyc) / (fy * (float)H_F), 0.0f);
        out_scores[p] = ds + hm;   // ALPHA = 1
    }
}

// ---------------- 6) 4x4 mean-pool features -> tokens [BS,480,1024] ----------------
__global__ __launch_bounds__(256)
void pool_tokens_kernel(const float* __restrict__ feat, float* __restrict__ tokens) {
    const int idx = blockIdx.x * 256 + threadIdx.x;
    if (idx >= BS * N_TOK * C_IN) return;
    const int c     = idx & 1023;
    const int token = (idx >> 10) % N_TOK;
    const int b     = idx / (N_TOK * C_IN);
    const int th = token / TOK_W;
    const int tw = token - th * TOK_W;
    float s = 0.f;
#pragma unroll
    for (int r = 0; r < 4; ++r) {
        const float* row = feat + (((size_t)b * C_IN + c) * H_F + th * 4 + r) * W_F + tw * 4;
        s += row[0] + row[1] + row[2] + row[3];
    }
    tokens[idx] = s * (1.0f / 16.0f);
}

// ---------------- 7) token scores (pool combined) + positions ----------------
__global__ __launch_bounds__(256)
void pool_scores_pos_kernel(const float* __restrict__ combined,
                            float* __restrict__ tok_scores,
                            float* __restrict__ out_pos) {
    const int idx = blockIdx.x * 256 + threadIdx.x;
    if (idx >= BS * N_TOK) return;
    const int token = idx % N_TOK;
    const int b     = idx / N_TOK;
    const int th = token / TOK_W;
    const int tw = token - th * TOK_W;
    float s = 0.f;
#pragma unroll
    for (int r = 0; r < 4; ++r) {
        const float* row = combined + ((size_t)b * H_F + th * 4 + r) * W_F + tw * 4;
        s += row[0] + row[1] + row[2] + row[3];
    }
    tok_scores[idx] = s * (1.0f / 16.0f);
    out_pos[(size_t)idx * 2 + 0] = (float)(th * 4 + 2);
    out_pos[(size_t)idx * 2 + 1] = (float)(tw * 4 + 2);
}

// ---------------- 8) top-100 of 480 (rank-based, matches top_k order) + gather ----------------
__global__ __launch_bounds__(512)
void topk_gather_kernel(const float* __restrict__ tok_scores,
                        const float* __restrict__ tokens,
                        float* __restrict__ centers) {
    const int b = blockIdx.x;
    __shared__ float s[N_TOK];
    __shared__ int sel[K_CLUST];
    for (int i = threadIdx.x; i < N_TOK; i += blockDim.x)
        s[i] = tok_scores[b * N_TOK + i];
    __syncthreads();
    for (int i = threadIdx.x; i < N_TOK; i += blockDim.x) {
        const float si = s[i];
        int rank = 0;
        for (int j = 0; j < N_TOK; ++j) {
            const float sj = s[j];
            rank += (sj > si) || (sj == si && j < i);
        }
        if (rank < K_CLUST) sel[rank] = i;
    }
    __syncthreads();
    for (int idx = threadIdx.x; idx < K_CLUST * C_IN; idx += blockDim.x) {
        const int r = idx >> 10;
        const int c = idx & 1023;
        centers[((size_t)b * K_CLUST + r) * C_IN + c] =
            tokens[((size_t)b * N_TOK + sel[r]) * C_IN + c];
    }
}

// ---------------- launch ----------------
extern "C" void kernel_launch(void* const* d_in, const int* in_sizes, int n_in,
                              void* d_out, int out_size, void* d_ws, size_t ws_size,
                              hipStream_t stream) {
    const float* features = (const float*)d_in[0];
    const float* calibs   = (const float*)d_in[1];
    const float* W1       = (const float*)d_in[2];
    const float* b1       = (const float*)d_in[3];
    const float* gn_gamma = (const float*)d_in[4];
    const float* gn_beta  = (const float*)d_in[5];
    const float* W2       = (const float*)d_in[6];
    const float* b2       = (const float*)d_in[7];

    // workspace layout (all offsets 256B aligned)
    char* ws = (char*)d_ws;
    const size_t fpad_bytes = (size_t)BS * HP * WP * C_IN * sizeof(_Float16); // 132,710,400
    const size_t wgt_bytes  = (size_t)9 * C_IN * OC * sizeof(_Float16);       //   4,718,592
    const size_t conv_bytes = (size_t)M_TOT * OC * sizeof(float);             //  62,914,560
    _Float16* fpad   = (_Float16*)ws;
    _Float16* wgt    = (_Float16*)(ws + fpad_bytes);
    float*    conv   = (float*)(ws + fpad_bytes + wgt_bytes);
    float*    stats  = (float*)(ws + fpad_bytes + wgt_bytes + conv_bytes);
    float*    tscore = stats + 2 * BS * 32;

    // output partitions (reference tuple, flattened in return order)
    float* out_scores  = (float*)d_out;                      // [8,48,160]
    float* out_centers = out_scores + (size_t)M_TOT;         // [8,100,1024]
    float* out_tokens  = out_centers + (size_t)BS * K_CLUST * C_IN; // [8,480,1024]
    float* out_pos     = out_tokens + (size_t)BS * N_TOK * C_IN;    // [8,480,2]

    {   // 1) pad+convert features
        const long long total = (long long)BS * HP * WP * C_IN;
        const int grid = (int)((total + 255) / 256);
        pad_f16_kernel<<<grid, 256, 0, stream>>>(features, fpad);
    }
    {   // 2) convert weights
        const int total = 9 * C_IN * OC;
        wconv_f16_kernel<<<(total + 255) / 256, 256, 0, stream>>>(W1, wgt);
    }
    {   // 3) WMMA conv: 1920 blocks x 32-row M-tiles
        conv3x3_wmma_kernel<<<M_TOT / 32, 128, 0, stream>>>(fpad, wgt, b1, conv);
    }
    {   // 4) GroupNorm stats: 8 batches x 32 groups
        gn_stats_kernel<<<BS * 32, 256, 0, stream>>>(conv, stats);
    }
    {   // 5) fused GN/ReLU/1x1/sigmoid/depth -> combined scores
        combined_kernel<<<M_TOT / 8, 256, 0, stream>>>(conv, stats, gn_gamma, gn_beta,
                                                       W2, b2, calibs, out_scores);
    }
    {   // 6) tokens
        const int total = BS * N_TOK * C_IN;
        pool_tokens_kernel<<<(total + 255) / 256, 256, 0, stream>>>(features, out_tokens);
    }
    {   // 7) token scores + positions
        const int total = BS * N_TOK;
        pool_scores_pos_kernel<<<(total + 255) / 256, 256, 0, stream>>>(out_scores, tscore, out_pos);
    }
    {   // 8) top-100 + center gather
        topk_gather_kernel<<<BS, 512, 0, stream>>>(tscore, out_tokens, out_centers);
    }
}
